// MPNN_32452772888838
// MI455X (gfx1250) — matrix operand order repaired and checked by reading, not compiled, not run on hardware
//
#include <hip/hip_runtime.h>
#include <hip/hip_bf16.h>

typedef __attribute__((ext_vector_type(16))) __bf16 v16bf;
typedef __attribute__((ext_vector_type(8)))  float  v8f;

#define NN 50000
#define EE 800000

// ---------------------------------------------------------------------------
// WMMA helpers (gfx1250: v_wmma_f32_16x16x32_bf16, D = A(16x32) * B(32x16) + C)
// ---------------------------------------------------------------------------
__device__ __forceinline__ v8f wmma_bf16(v16bf a, v16bf b, v8f c) {
  return __builtin_amdgcn_wmma_f32_16x16x32_bf16(
      /*neg_a=*/false, a, /*neg_b=*/false, b,
      /*c_mod=*/(short)0, c, /*reuse_a=*/false, /*reuse_b=*/false);
}

// A tile: 16 rows x 32 K, bf16, from row-major LDS [rows][ld].
// Lanes 0-15: row=lane, elems 0..7 = K kb..kb+7, elems 8..15 = K kb+16..kb+23
// (kb = k0); lanes 16-31: same rows, kb = k0+8.  Two ds_load_b128 per lane.
__device__ __forceinline__ v16bf load_a_lds(const __bf16* As, int ld, int k0) {
  const int lane = threadIdx.x & 31;
  const int row  = lane & 15;
  const int kb   = k0 + ((lane >> 4) << 3);
  const uint4* p = (const uint4*)(As + row * ld + kb);  // 16B aligned
  union { v16bf v; uint4 q[2]; } u;
  u.q[0] = p[0];   // K kb .. kb+7
  u.q[1] = p[2];   // K kb+16 .. kb+23
  return u.v;
}

// B tile from pre-swizzled fragment-major weights:
//   Wp[kt][nt][lane][16]  (kt = k0/32, nt = n0/16, ntiles = N/16)
// One contiguous 32-byte load per lane.
__device__ __forceinline__ v16bf load_b_packed(const __bf16* Wp, int ntiles,
                                               int kt, int nt) {
  const int lane = threadIdx.x & 31;
  return ((const v16bf*)Wp)[((size_t)(kt * ntiles + nt) * 32) + lane];
}

__device__ __forceinline__ void fadd_atomic(float* p, float v) {
  __hip_atomic_fetch_add(p, v, __ATOMIC_RELAXED, __HIP_MEMORY_SCOPE_AGENT);
}

__device__ __forceinline__ unsigned pack2bf(float a, float b) {
  union { __bf16 h[2]; unsigned u; } x;
  x.h[0] = (__bf16)a; x.h[1] = (__bf16)b;
  return x.u;
}

// ---------------------------------------------------------------------------
// fp32 -> bf16 plain conversion (node features)
// ---------------------------------------------------------------------------
__global__ void cvt_f32_bf16(const float* __restrict__ in, __bf16* __restrict__ out, int n) {
  int i = blockIdx.x * blockDim.x + threadIdx.x;
  if (i < n) out[i] = (__bf16)in[i];
}

// ---------------------------------------------------------------------------
// fp32 [K][N] row-major -> bf16 WMMA-B fragment-major [K/32][N/16][32 lanes][16]
// lane = (n%16) | (((k%32)>>3)&1)<<4 ; elem j = (k%8) | ((k%32)>>4)<<3  (per ISA 16-bit B layout)
// ---------------------------------------------------------------------------
__global__ void pack_b_kernel(const float* __restrict__ in, __bf16* __restrict__ out,
                              int K, int N) {
  int i = blockIdx.x * blockDim.x + threadIdx.x;
  if (i >= K * N) return;
  const int k = i / N, n = i % N;
  const int kt = k >> 5, kr = k & 31;
  const int nt = n >> 4, nr = n & 15;
  const int lane = nr | (((kr >> 3) & 1) << 4);
  const int j = (kr & 7) | ((kr >> 4) << 3);
  out[(((size_t)(kt * (N >> 4) + nt) * 32) + lane) * 16 + j] = (__bf16)in[i];
}

// ---------------------------------------------------------------------------
// Per-edge message MLP, fused with scatter-add aggregation.
// Block = 256 threads (8 waves) = 32 edges (two 16-row tiles; B reused).
//   H1 = relu([x_dst|x_src] @ W1[0:2F] + e * W1[2F] + b1)   (32 x 128)
//   msg = H1 @ W2 + b2                                       (32 x 256)
//   aggr[dst] += msg   (fp32 global atomics, lands in L2)
// ---------------------------------------------------------------------------
template <int F>
__global__ __launch_bounds__(256) void msg_mlp_kernel(
    const __bf16* __restrict__ feat,    // [NN, F] bf16
    const float*  __restrict__ eattr,   // [EE]
    const int*    __restrict__ srcI,    // [EE]
    const int*    __restrict__ dstI,    // [EE]
    const __bf16* __restrict__ W1p,     // packed [2F/32][8][32][16]
    const float*  __restrict__ W1last,  // fp32 row 2F of m1_w1 (edge_attr row)
    const float*  __restrict__ b1,      // [128]
    const __bf16* __restrict__ W2p,     // packed [128/32][16][32][16]
    const float*  __restrict__ b2,      // [256]
    float* __restrict__ aggr)           // [NN, 256]
{
  constexpr int K1 = 2 * F;
  __shared__ __bf16 Ain[32][K1];
  __shared__ __bf16 H1s[32][128];
  __shared__ float  eSh[32];
  __shared__ int    dSh[32];

  const int tid = threadIdx.x;
  const int e0  = blockIdx.x * 32;

  if (tid < 32) {
    eSh[tid] = eattr[e0 + tid];
    dSh[tid] = dstI[e0 + tid];
  }
  {  // gather: thread t copies a contiguous F/4 chunk of row (t>>3), 128-bit ops
    const int m   = tid >> 3;
    const int off = (tid & 7) * (K1 / 8);  // F/4 divides F: no dst/src straddle
    const int d = dstI[e0 + m], s = srcI[e0 + m];
    const __bf16* p = (off < F) ? (feat + (size_t)d * F + off)
                                : (feat + (size_t)s * F + (off - F));
    const uint4* s4 = (const uint4*)p;
    uint4* d4 = (uint4*)&Ain[m][off];
#pragma unroll
    for (int j = 0; j < K1 / 64; ++j) d4[j] = s4[j];
  }
  __syncthreads();

  const int wave = tid >> 5;
  const int lane = tid & 31;

  // ---- stage 1: H1 (128 cols -> 1 col tile per wave, 2 row tiles), K = 2F ----
  {
    const int n0 = wave * 16;
    v8f c0 = {}, c1 = {};
#pragma unroll
    for (int k0 = 0; k0 < K1; k0 += 32) {
      const v16bf b = load_b_packed(W1p, 8, k0 >> 5, wave);
      c0 = wmma_bf16(load_a_lds(&Ain[0][0],  K1, k0), b, c0);
      c1 = wmma_bf16(load_a_lds(&Ain[16][0], K1, k0), b, c1);
    }
    const int   col   = n0 + (lane & 15);
    const float wlast = W1last[col];
    const float bias  = b1[col];
#pragma unroll
    for (int v = 0; v < 8; ++v) {
      const int m = ((lane >> 4) << 3) + v;
      H1s[m][col]      = (__bf16)fmaxf(c0[v] + eSh[m] * wlast + bias, 0.0f);
      H1s[m + 16][col] = (__bf16)fmaxf(c1[v] + eSh[m + 16] * wlast + bias, 0.0f);
    }
  }
  __syncthreads();

  // ---- stage 2: msg (256 cols -> 2 col tiles per wave, 2 row tiles), K = 128 ----
#pragma unroll
  for (int t = 0; t < 2; ++t) {
    const int n0 = wave * 32 + t * 16;
    v8f c0 = {}, c1 = {};
#pragma unroll
    for (int k0 = 0; k0 < 128; k0 += 32) {
      const v16bf b = load_b_packed(W2p, 16, k0 >> 5, n0 >> 4);
      c0 = wmma_bf16(load_a_lds(&H1s[0][0],  128, k0), b, c0);
      c1 = wmma_bf16(load_a_lds(&H1s[16][0], 128, k0), b, c1);
    }
    const int   col  = n0 + (lane & 15);
    const float bias = b2[col];
#pragma unroll
    for (int v = 0; v < 8; ++v) {
      const int m = ((lane >> 4) << 3) + v;
      fadd_atomic(&aggr[(size_t)dSh[m] * 256 + col],      c0[v] + bias);
      fadd_atomic(&aggr[(size_t)dSh[m + 16] * 256 + col], c1[v] + bias);
    }
  }
}

// ---------------------------------------------------------------------------
// Node update: h = relu([feat | aggr] @ Uw + Ub)   (16 nodes / block)
// REDUCE=false: store h as bf16 via LDS (coalesced 128-bit stores)
// REDUCE=true : fused graph pooling (block LDS reduce + atomic per column)
// ---------------------------------------------------------------------------
template <int F, bool REDUCE>
__global__ __launch_bounds__(256) void update_kernel(
    const __bf16* __restrict__ feat,   // [NN, F] bf16
    const float*  __restrict__ aggr,   // [NN, 256] fp32
    const __bf16* __restrict__ Uwp,    // packed [(F+256)/32][16][32][16]
    const float*  __restrict__ Ub,     // [256]
    __bf16* __restrict__ hOut,         // [NN, 256] (if !REDUCE)
    float*  __restrict__ g)            // [256]     (if REDUCE)
{
  constexpr int K = F + 256;
  __shared__ __bf16 Ain[16][K];
  __shared__ float  Dsh[REDUCE ? 16 : 1][REDUCE ? 256 : 1];
  __shared__ __bf16 Hb[REDUCE ? 1 : 16][REDUCE ? 1 : 256];

  const int tid = threadIdx.x;
  const int r0  = blockIdx.x * 16;  // 50000 % 16 == 0

  {  // feature part of A (bf16, 128-bit copies)
    const int m   = tid >> 4;
    const int off = (tid & 15) * (F / 16);
    const uint4* s4 = (const uint4*)(feat + (size_t)(r0 + m) * F + off);
    uint4* d4 = (uint4*)&Ain[m][off];
#pragma unroll
    for (int j = 0; j < F / 128; ++j) d4[j] = s4[j];
  }
  {  // aggr part of A: fp32 -> bf16, packed 128-bit stores
    const int m   = tid >> 4;
    const int off = (tid & 15) * 16;
    const float4* p = (const float4*)(aggr + (size_t)(r0 + m) * 256 + off);
    unsigned w[8];
#pragma unroll
    for (int q = 0; q < 4; ++q) {
      const float4 f = p[q];
      w[2 * q]     = pack2bf(f.x, f.y);
      w[2 * q + 1] = pack2bf(f.z, f.w);
    }
    uint4* d4 = (uint4*)&Ain[m][F + off];
    d4[0] = make_uint4(w[0], w[1], w[2], w[3]);
    d4[1] = make_uint4(w[4], w[5], w[6], w[7]);
  }
  __syncthreads();

  const int wave = tid >> 5;
  const int lane = tid & 31;

#pragma unroll
  for (int t = 0; t < 2; ++t) {
    const int n0 = wave * 32 + t * 16;
    v8f c = {};
#pragma unroll
    for (int k0 = 0; k0 < K; k0 += 32)
      c = wmma_bf16(load_a_lds(&Ain[0][0], K, k0),
                    load_b_packed(Uwp, 16, k0 >> 5, n0 >> 4), c);
    const int   col  = n0 + (lane & 15);
    const float bias = Ub[col];
#pragma unroll
    for (int v = 0; v < 8; ++v) {
      const int m = ((lane >> 4) << 3) + v;
      const float val = fmaxf(c[v] + bias, 0.0f);
      if (REDUCE) Dsh[m][col] = val;
      else        Hb[m][col]  = (__bf16)val;
    }
  }
  __syncthreads();

  if (REDUCE) {
    float s = 0.0f;
#pragma unroll
    for (int m = 0; m < 16; ++m) s += Dsh[m][tid];
    fadd_atomic(&g[tid], s);
  } else {
    // coalesced 128-bit store of the h tile
    const int m   = tid >> 4;
    const int off = (tid & 15) * 16;
    const uint4* s4 = (const uint4*)&Hb[m][off];
    uint4* d4 = (uint4*)(hOut + (size_t)(r0 + m) * 256 + off);
    d4[0] = s4[0];
    d4[1] = s4[1];
  }
}

// ---------------------------------------------------------------------------
// Readout: out = relu(g @ r_w1 + r_b1) @ r_w2 + r_b2   (1x256 -> 1x10, tiny)
// ---------------------------------------------------------------------------
__global__ void readout_kernel(const float* __restrict__ g,
                               const float* __restrict__ w1, const float* __restrict__ b1,
                               const float* __restrict__ w2, const float* __restrict__ b2,
                               float* __restrict__ out)
{
  __shared__ float h[128];
  const int t = threadIdx.x;
  if (t < 128) {
    float s = b1[t];
    for (int k = 0; k < 256; ++k) s += g[k] * w1[k * 128 + t];
    h[t] = fmaxf(s, 0.0f);
  }
  __syncthreads();
  if (t < 10) {
    float s = b2[t];
    for (int k = 0; k < 128; ++k) s += h[k] * w2[k * 10 + t];
    out[t] = s;
  }
}

// ---------------------------------------------------------------------------
extern "C" void kernel_launch(void* const* d_in, const int* in_sizes, int n_in,
                              void* d_out, int out_size, void* d_ws, size_t ws_size,
                              hipStream_t stream) {
  (void)in_sizes; (void)n_in; (void)out_size; (void)ws_size;

  const float* x      = (const float*)d_in[0];
  const float* eattr  = (const float*)d_in[1];
  const float* m1_w1  = (const float*)d_in[2];
  const float* m1_b1  = (const float*)d_in[3];
  const float* m1_w2  = (const float*)d_in[4];
  const float* m1_b2  = (const float*)d_in[5];
  const float* u1_w   = (const float*)d_in[6];
  const float* u1_b   = (const float*)d_in[7];
  const float* m2_w1  = (const float*)d_in[8];
  const float* m2_b1  = (const float*)d_in[9];
  const float* m2_w2  = (const float*)d_in[10];
  const float* m2_b2  = (const float*)d_in[11];
  const float* u2_w   = (const float*)d_in[12];
  const float* u2_b   = (const float*)d_in[13];
  const float* r_w1   = (const float*)d_in[14];
  const float* r_b1   = (const float*)d_in[15];
  const float* r_w2   = (const float*)d_in[16];
  const float* r_b2   = (const float*)d_in[17];
  const int*   ei     = (const int*)d_in[18];
  const int*   srcI   = ei;
  const int*   dstI   = ei + EE;
  float*       out    = (float*)d_out;

  // ---- workspace carve-up (~91 MB, all chunks 256B-aligned) ----
  char* ws = (char*)d_ws;
  float*  aggr  = (float*)ws;  ws += (size_t)NN * 256 * 4;  // reused both layers
  __bf16* xb    = (__bf16*)ws; ws += (size_t)NN * 128 * 2;
  __bf16* h1b   = (__bf16*)ws; ws += (size_t)NN * 256 * 2;
  __bf16* wm1a  = (__bf16*)ws; ws += (size_t)256 * 128 * 2;  // packed, rows 0..255
  __bf16* wm1b  = (__bf16*)ws; ws += (size_t)128 * 256 * 2;  // packed
  __bf16* wu1   = (__bf16*)ws; ws += (size_t)384 * 256 * 2;  // packed
  __bf16* wm2a  = (__bf16*)ws; ws += (size_t)512 * 128 * 2;  // packed, rows 0..511
  __bf16* wm2b  = (__bf16*)ws; ws += (size_t)128 * 256 * 2;  // packed
  __bf16* wu2   = (__bf16*)ws; ws += (size_t)512 * 256 * 2;  // packed
  float*  g     = (float*)ws;  ws += 256 * 4;

  hipMemsetAsync(aggr, 0, (size_t)NN * 256 * 4, stream);
  hipMemsetAsync(g, 0, 256 * 4, stream);

  {  // node features -> bf16
    const int n = NN * 128;
    cvt_f32_bf16<<<(n + 255) / 256, 256, 0, stream>>>(x, xb, n);
  }
  auto pack = [&](const float* s, __bf16* d, int K, int N) {
    const int n = K * N;
    pack_b_kernel<<<(n + 255) / 256, 256, 0, stream>>>(s, d, K, N);
  };
  pack(m1_w1, wm1a, 256, 128);   // last row (edge_attr) handled in fp32 fixup
  pack(m1_w2, wm1b, 128, 256);
  pack(u1_w,  wu1,  384, 256);
  pack(m2_w1, wm2a, 512, 128);
  pack(m2_w2, wm2b, 128, 256);
  pack(u2_w,  wu2,  512, 256);

  // ---- layer 1 ----
  msg_mlp_kernel<128><<<EE / 32, 256, 0, stream>>>(
      xb, eattr, srcI, dstI, wm1a, m1_w1 + (size_t)256 * 128, m1_b1, wm1b, m1_b2, aggr);
  update_kernel<128, false><<<NN / 16, 256, 0, stream>>>(
      xb, aggr, wu1, u1_b, h1b, nullptr);

  // ---- layer 2 (reuse aggr buffer) ----
  hipMemsetAsync(aggr, 0, (size_t)NN * 256 * 4, stream);
  msg_mlp_kernel<256><<<EE / 32, 256, 0, stream>>>(
      h1b, eattr, srcI, dstI, wm2a, m2_w1 + (size_t)512 * 128, m2_b1, wm2b, m2_b2, aggr);
  update_kernel<256, true><<<NN / 16, 256, 0, stream>>>(
      h1b, aggr, wu2, u2_b, nullptr, g);

  // ---- readout ----
  readout_kernel<<<1, 128, 0, stream>>>(g, r_w1, r_b1, r_w2, r_b2, out);
}
